// MIKT_34892314312894
// MI455X (gfx1250) — matrix operand order repaired and checked by reading, not compiled, not run on hardware
//
#include <hip/hip_runtime.h>

#define PMAX 20000
#define KD 256
#define DD 128
#define BB 128
#define SL 200

typedef __bf16 bf16;
typedef __attribute__((ext_vector_type(2)))  __bf16 bf16x2;
typedef __attribute__((ext_vector_type(16))) __bf16 v16bf;
typedef __attribute__((ext_vector_type(8)))  float  v8f;

struct Q2 { uint4 a, b; };          // 32B raw fragment
struct P8 { bf16x2 v[8]; };         // 8 packed bf16 pairs = 32B

__device__ __forceinline__ float sgm(float x){ return 1.f/(1.f+__expf(-x)); }

__device__ __forceinline__ v8f wmma_bf(v16bf a, v16bf b, v8f c){
  // D = A(16x32 bf16) * B(32x16 bf16) + C(16x16 f32)
  return __builtin_amdgcn_wmma_f32_16x16x32_bf16(false, a, false, b, (short)0, c, false, false);
}

// B fragment / bf16 A fragment: two 16B LDS loads, no repacking moves
__device__ __forceinline__ v16bf ld_frag(const bf16* p0, const bf16* p1){
  Q2 q;
  q.a = *(const uint4*)p0;
  q.b = *(const uint4*)p1;
  return __builtin_bit_cast(v16bf, q);
}

// A fragment from f32 LDS state: four b128 loads -> v_cvt_pk_bf16_f32 packs
__device__ __forceinline__ v16bf cvt_frag(const float* p0){
  float4 q0 = *(const float4*)(p0);
  float4 q1 = *(const float4*)(p0+4);
  float4 q2 = *(const float4*)(p0+16);
  float4 q3 = *(const float4*)(p0+20);
  P8 r;
  r.v[0]=bf16x2{(__bf16)q0.x,(__bf16)q0.y};
  r.v[1]=bf16x2{(__bf16)q0.z,(__bf16)q0.w};
  r.v[2]=bf16x2{(__bf16)q1.x,(__bf16)q1.y};
  r.v[3]=bf16x2{(__bf16)q1.z,(__bf16)q1.w};
  r.v[4]=bf16x2{(__bf16)q2.x,(__bf16)q2.y};
  r.v[5]=bf16x2{(__bf16)q2.z,(__bf16)q2.w};
  r.v[6]=bf16x2{(__bf16)q3.x,(__bf16)q3.y};
  r.v[7]=bf16x2{(__bf16)q3.z,(__bf16)q3.w};
  return __builtin_bit_cast(v16bf, r);
}

// ---------------------------------------------------------------------------
// Setup: transpose weights for coalesced per-thread dots, build bf16 B-operands
// ---------------------------------------------------------------------------
__global__ void k_prep(const float* af_w1, const float* af_w2, const float* sf_w1,
                       const float* sf_w2, const float* pr_w, const float* pa_w1,
                       const float* no_w1, const float* no_w2, const float* ao_w,
                       const float* pl_w, const float* pc_w,
                       float* afw1T, float* afw2T, float* sfcT, float* prT, float* paT,
                       float* no1T, float* no2T, float* aoT, float* plT, float* pcT,
                       bf16* w1t, bf16* w2t)
{
  int gt = blockIdx.x*blockDim.x + threadIdx.x;
  int gs = gridDim.x*blockDim.x;
  for (int i=gt; i<DD*2*DD; i+=gs){ int r=i/(2*DD), c=i%(2*DD); afw1T[c*DD+r]=af_w1[i]; }
  for (int i=gt; i<DD*DD; i+=gs){
    int r=i/DD, c=i%DD;
    afw2T[c*DD+r]=af_w2[i];
    no1T [c*DD+r]=no_w1[i];
    no2T [c*DD+r]=no_w2[i];
    aoT  [c*DD+r]=ao_w[i];
    plT  [c*DD+r]=pl_w[i];
    pcT  [c*DD+r]=pc_w[i];
    sfcT [c*DD+r]=sf_w1[r*3*DD + 2*DD + c];       // eff block of sf_w1
    w1t  [r*DD+c]=(bf16)sf_w1[r*3*DD + c];        // B-layout [N][K] = sf_w1 (state block)
    w2t  [r*DD+c]=(bf16)sf_w2[i];                 // B-layout [N][K] = sf_w2
  }
  for (int i=gt; i<DD*3*DD; i+=gs){
    int r=i/(3*DD), c=i%(3*DD);
    prT[c*DD+r]=pr_w[i];
    paT[c*DD+r]=pa_w1[i];
  }
}

// time_proj[t][d] = sum_j time_state[t][j] * sf_w1[d][D + j]   (stg block, 200 rows)
__global__ void k_timeproj(const float* time_state, const float* sf_w1, bf16* tp){
  int t=blockIdx.x, d=threadIdx.x;
  float acc=0.f;
  for (int j=0;j<DD;j++) acc += time_state[t*DD+j]*sf_w1[d*3*DD + DD + j];
  tp[t*DD+d]=(bf16)acc;
}

// q_skill = skill_embed @ sl_w^T + sl_b
__global__ void k_qskill(const float* skill_embed, const float* sl_w, const float* sl_b,
                         float* q_skill){
  int k=blockIdx.x, d=threadIdx.x;
  float acc=sl_b[d];
  for (int j=0;j<DD;j++) acc += skill_embed[k*DD+j]*sl_w[d*DD+j];
  q_skill[k*DD+d]=acc;
}

// ---------------------------------------------------------------------------
// Static problem embedding: pro_e[p] and sigmoid(diff)
// ---------------------------------------------------------------------------
__global__ void __launch_bounds__(128) k_pro(
    const float* pro_embed, const float* skill_embed, const float* pl_b,
    const float* pc_b, const float* pro_diff_w, const int* pro2skill,
    const float* plT, const float* pcT, const float* q_skill,
    float* pro_e, float* diff_sig)
{
  __shared__ float xr[DD], qp[DD], at[KD], smn[DD], rb[DD];
  int p=blockIdx.x, d=threadIdx.x;
  xr[d]=pro_embed[(size_t)p*DD+d];
  __syncthreads();
  { float acc=pl_b[d];
    for (int j=0;j<DD;j++) acc += xr[j]*plT[j*DD+d];
    qp[d]=acc; }
  __syncthreads();
  for (int kk=d; kk<KD; kk+=DD){
    float s=0.f;
    for (int j=0;j<DD;j++) s += qp[j]*q_skill[kk*DD+j];
    s *= 0.08838834764831845f;                     // 1/sqrt(128)
    if (pro2skill[(size_t)p*KD+kk]==0) s=-1e9f;
    at[kk]=s;
  }
  __syncthreads();
  rb[d]=fmaxf(at[d],at[d+DD]); __syncthreads();
  for (int off=64; off; off>>=1){ if(d<off) rb[d]=fmaxf(rb[d],rb[d+off]); __syncthreads(); }
  float mx=rb[0]; __syncthreads();
  float e0=__expf(at[d]-mx), e1=__expf(at[d+DD]-mx);
  at[d]=e0; at[d+DD]=e1; rb[d]=e0+e1; __syncthreads();
  for (int off=64; off; off>>=1){ if(d<off) rb[d]+=rb[d+off]; __syncthreads(); }
  float inv=1.f/rb[0]; __syncthreads();
  at[d]*=inv; at[d+DD]*=inv;
  float sm=0.f, cnt=0.f;
  for (int k=0;k<KD;k++){
    float w=(float)pro2skill[(size_t)p*KD+k];
    sm  += w*skill_embed[k*DD+d];
    cnt += w;
  }
  smn[d]=sm/(cnt+1e-8f);
  __syncthreads();
  float pcv=pc_b[d];
  for (int j=0;j<DD;j++) pcv += smn[j]*pcT[j*DD+d];
  float pe_=0.f;
  for (int k=0;k<KD;k++) pe_ += at[k]*skill_embed[k*DD+d];
  float dsig=sgm(pro_diff_w[p]);
  pro_e[(size_t)p*DD+d]= pe_ + dsig*pcv;
  if (d==0) diff_sig[p]=dsig;
}

// gather per-position rasch / X / diff
__global__ void k_gather(const int* next_problem, const int* next_ans,
                         const float* pro_e, const float* ans_emb, const float* diff_sig,
                         float* rasch, float* Xg, float* diff_bs){
  int i=blockIdx.x, d=threadIdx.x;
  int p=next_problem[i], a=next_ans[i];
  float r=pro_e[(size_t)p*DD+d];
  rasch[(size_t)i*DD+d]=r;
  Xg[(size_t)i*DD+d]=r + ans_emb[a*DD+d];
  if (d==0) diff_bs[i]=diff_sig[p];
}

// ---------------------------------------------------------------------------
// Sequential scan: 1 workgroup (8 wave32) per batch element, state in LDS.
// Per step: two [256,128]x[128,128] WMMA bf16 GEMMs + small VALU MLPs.
// ---------------------------------------------------------------------------
__global__ void __launch_bounds__(256) k_scan(
    const float* skill_state0, const float* all_state0, const float* time_state,
    const float* af_b1, const float* af_b2, const float* sf_b1, const float* sf_b2,
    const float* pr_b, const float* pa_b1, const float* pa_w2, const float* pa_b2,
    const float* no_b1, const float* no_b2, const float* ao_b,
    const float* afw1T, const float* afw2T, const float* sfcT,
    const float* prT, const float* paT, const float* no1T, const float* no2T,
    const float* aoT,
    const bf16* w1t, const bf16* w2t, const bf16* tp,
    const int* pro2skill, const int* next_problem,
    const float* rasch, const float* Xg, const float* diff_bs,
    float* out)
{
  extern __shared__ char smem[];
  float* ss  = (float*)smem;                  // [256][128] f32 state   (128KB)
  bf16*  m1  = (bf16*)(smem + 131072);        // [256][128] bf16 relu   ( 64KB)
  bf16*  w1s = (bf16*)(smem + 196608);        // B-operand W1           ( 32KB)
  bf16*  w2s = (bf16*)(smem + 229376);        // B-operand W2           ( 32KB)
  float* fb  = (float*)(smem + 262144);
  float *h1=fb, *fga=fb+128, *rowc=fb+256, *pe=fb+384, *Xs=fb+512, *as_=fb+640,
        *ag=fb+768, *nd=fb+896, *n2a=fb+1024, *n2b=fb+1152, *hpa=fb+1280,
        *t1=fb+1408, *tg=fb+1536, *sb2=fb+1664, *sc=fb+1792, *rb=fb+2048;
  int *gaps=(int*)(fb+2304), *ltm=(int*)(fb+2560), *npk=(int*)(fb+2816);
  int*   misc_i=(int*)(fb+3072);
  float* misc_f=(float*)(fb+3072);

  const int tid=threadIdx.x, b=blockIdx.x;
  const int wid=tid>>5, lane=tid&31, lm=lane&15, lh=lane>>4;

  for (int i=tid; i<KD*DD; i+=256) ss[i]=skill_state0[i];
  for (int i=tid; i<DD*DD; i+=256){ w1s[i]=w1t[i]; w2s[i]=w2t[i]; }
  if (tid<DD){ as_[tid]=all_state0[tid]; ag[tid]=time_state[DD+tid]; sb2[tid]=sf_b2[tid]; }
  ltm[tid]=0;
  if (b==0 && tid==0) out[BB*SL]=0.0f;        // aux scalar output
  __syncthreads();

  for (int t=0; t<SL; ++t){
    const int bs=b*SL+t;
    if (tid<DD){ pe[tid]=rasch[(size_t)bs*DD+tid]; Xs[tid]=Xg[(size_t)bs*DD+tid]; }
    if (tid==0){ misc_i[0]=next_problem[bs]; misc_f[1]=diff_bs[bs]; }
    if (t+1<SL && tid<4){                      // global_prefetch_b8 next step rows
      __builtin_prefetch(&rasch[(size_t)(bs+1)*DD + tid*32], 0, 3);
      __builtin_prefetch(&Xg   [(size_t)(bs+1)*DD + tid*32], 0, 3);
    }
    __syncthreads();
    const int pro=misc_i[0];
    const float diff=misc_f[1];

    // gaps / skill mask + forget-gate hidden
    { int v=pro2skill[(size_t)pro*KD+tid];
      npk[tid]=v;
      gaps[tid]= t - (v ? ltm[tid] : 0); }
    if (tid<DD){
      float acc=af_b1[tid];
      for (int j=0;j<DD;j++) acc += as_[j]*afw1T[j*DD+tid];
      for (int j=0;j<DD;j++) acc += ag[j]*afw1T[(DD+j)*DD+tid];
      h1[tid]=fmaxf(acc,0.f);
    }
    __syncthreads();
    if (tid<DD){
      float acc=af_b2[tid];
      for (int j=0;j<DD;j++) acc += h1[j]*afw2T[j*DD+tid];
      fga[tid]= as_[tid]*sgm(acc);              // forget_all
    }
    __syncthreads();
    if (tid<DD){                                 // eff-block contribution + bias
      float acc=sf_b1[tid];
      for (int j=0;j<DD;j++) acc += fga[j]*sfcT[j*DD+tid];
      rowc[tid]=acc;
    }
    __syncthreads();

    // ---- GEMM1: M1 = relu(ss @ W1^T + time_proj[gap] + rowc) -> bf16 ----
    {
      const int mt0=wid*2, mt1=wid*2+1;
      const int ar0=mt0*16+lm, ar1=mt1*16+lm;
      v16bf a0[4], a1[4];
      #pragma unroll
      for (int ks=0; ks<4; ks++){
        a0[ks]=cvt_frag(ss + ar0*DD + ks*32 + lh*8);
        a1[ks]=cvt_frag(ss + ar1*DD + ks*32 + lh*8);
      }
      #pragma unroll
      for (int ntp=0; ntp<4; ntp++){
        const int ntA=2*ntp, ntB=2*ntp+1;
        v16bf bA[4], bB[4];
        #pragma unroll
        for (int ks=0; ks<4; ks++){
          const bf16* bpA = w1s + (ntA*16+lm)*DD + ks*32 + lh*8;
          const bf16* bpB = w1s + (ntB*16+lm)*DD + ks*32 + lh*8;
          bA[ks]=ld_frag(bpA, bpA+16);
          bB[ks]=ld_frag(bpB, bpB+16);
        }
        v8f c00={}, c10={}, c01={}, c11={};
        #pragma unroll
        for (int ks=0; ks<4; ks++){           // 4 independent accumulator chains
          c00 = wmma_bf(a0[ks], bA[ks], c00);
          c10 = wmma_bf(a1[ks], bA[ks], c10);
          c01 = wmma_bf(a0[ks], bB[ks], c01);
          c11 = wmma_bf(a1[ks], bB[ks], c11);
        }
        const int nA=ntA*16+lm, nB=ntB*16+lm;
        const float rcA=rowc[nA], rcB=rowc[nB];
        #pragma unroll
        for (int r=0;r<8;r++){
          int m0=mt0*16+lh*8+r, m1r=mt1*16+lh*8+r;
          float v00=c00[r]+rcA+(float)tp[gaps[m0]*DD+nA];
          float v10=c10[r]+rcA+(float)tp[gaps[m1r]*DD+nA];
          float v01=c01[r]+rcB+(float)tp[gaps[m0]*DD+nB];
          float v11=c11[r]+rcB+(float)tp[gaps[m1r]*DD+nB];
          m1[m0*DD+nA]=(bf16)fmaxf(v00,0.f);
          m1[m1r*DD+nA]=(bf16)fmaxf(v10,0.f);
          m1[m0*DD+nB]=(bf16)fmaxf(v01,0.f);
          m1[m1r*DD+nB]=(bf16)fmaxf(v11,0.f);
        }
      }
    }
    __syncthreads();

    // ---- GEMM2: sf = sigmoid(M1 @ W2^T + b2); ss *= sf (mask->1) ----
    {
      const int mt0=wid*2, mt1=wid*2+1;
      const int ar0=mt0*16+lm, ar1=mt1*16+lm;
      v16bf a0[4], a1[4];
      #pragma unroll
      for (int ks=0; ks<4; ks++){
        const bf16* p0 = m1 + ar0*DD + ks*32 + lh*8;
        const bf16* p1 = m1 + ar1*DD + ks*32 + lh*8;
        a0[ks]=ld_frag(p0, p0+16);
        a1[ks]=ld_frag(p1, p1+16);
      }
      #pragma unroll
      for (int ntp=0; ntp<4; ntp++){
        const int ntA=2*ntp, ntB=2*ntp+1;
        v16bf bA[4], bB[4];
        #pragma unroll
        for (int ks=0; ks<4; ks++){
          const bf16* bpA = w2s + (ntA*16+lm)*DD + ks*32 + lh*8;
          const bf16* bpB = w2s + (ntB*16+lm)*DD + ks*32 + lh*8;
          bA[ks]=ld_frag(bpA, bpA+16);
          bB[ks]=ld_frag(bpB, bpB+16);
        }
        v8f c00={}, c10={}, c01={}, c11={};
        #pragma unroll
        for (int ks=0; ks<4; ks++){
          c00 = wmma_bf(a0[ks], bA[ks], c00);
          c10 = wmma_bf(a1[ks], bA[ks], c10);
          c01 = wmma_bf(a0[ks], bB[ks], c01);
          c11 = wmma_bf(a1[ks], bB[ks], c11);
        }
        const int nA=ntA*16+lm, nB=ntB*16+lm;
        const float bA_=sb2[nA], bB_=sb2[nB];
        #pragma unroll
        for (int r=0;r<8;r++){
          int m0=mt0*16+lh*8+r, m1r=mt1*16+lh*8+r;
          float s00 = npk[m0]  ? sgm(c00[r]+bA_) : 1.f;
          float s10 = npk[m1r] ? sgm(c10[r]+bA_) : 1.f;
          float s01 = npk[m0]  ? sgm(c01[r]+bB_) : 1.f;
          float s11 = npk[m1r] ? sgm(c11[r]+bB_) : 1.f;
          ss[m0*DD+nA]  *= s00;
          ss[m1r*DD+nA] *= s10;
          ss[m0*DD+nB]  *= s01;
          ss[m1r*DD+nB] *= s11;
        }
      }
    }
    __syncthreads();

    // ---- attention over skills: scores -> softmax -> need ----
    { float s=0.f; const float* row=ss+tid*DD;
      for (int d=0;d<DD;d++) s += pe[d]*row[d];
      s *= (1.f/DD);
      if (!npk[tid]) s=-1e9f;
      sc[tid]=s; }
    __syncthreads();
    rb[tid]=sc[tid]; __syncthreads();
    for (int off=128; off; off>>=1){ if(tid<off) rb[tid]=fmaxf(rb[tid],rb[tid+off]); __syncthreads(); }
    { float mx=rb[0]; __syncthreads();
      float e=__expf(sc[tid]-mx); sc[tid]=e; rb[tid]=e; }
    __syncthreads();
    for (int off=128; off; off>>=1){ if(tid<off) rb[tid]+=rb[tid+off]; __syncthreads(); }
    { float inv=1.f/rb[0]; __syncthreads(); sc[tid]*=inv; }
    __syncthreads();
    if (tid<DD){
      float acc=0.f;
      for (int k=0;k<KD;k++) acc += sc[k]*ss[k*DD+tid];
      nd[tid]=acc;
    }
    __syncthreads();

    // ---- pr gate -> need2 ----
    if (tid<DD){
      float acc=pr_b[tid];
      for (int j=0;j<DD;j++) acc += nd[j]*prT[j*DD+tid];
      for (int j=0;j<DD;j++) acc += fga[j]*prT[(DD+j)*DD+tid];
      for (int j=0;j<DD;j++) acc += pe[j]*prT[(2*DD+j)*DD+tid];
      float aa=sgm(acc);
      n2a[tid]=(1.f-aa)*nd[tid];
      n2b[tid]=aa*fga[tid];
    }
    __syncthreads();

    // ---- ability hidden + all_state update + no1 (independent) ----
    float asv=0.f;
    if (tid<DD){
      float acc=pa_b1[tid];
      for (int j=0;j<DD;j++) acc += n2a[j]*paT[j*DD+tid];
      for (int j=0;j<DD;j++) acc += n2b[j]*paT[(DD+j)*DD+tid];
      for (int j=0;j<DD;j++) acc += pe[j]*paT[(2*DD+j)*DD+tid];
      hpa[tid]=fmaxf(acc,0.f);
      float a1=ao_b[tid], a2=no_b1[tid];
      for (int j=0;j<DD;j++){ float x=Xs[j]; a1+=x*aoT[j*DD+tid]; a2+=x*no1T[j*DD+tid]; }
      asv = fga[tid] + tanhf(a1);
      t1[tid]=tanhf(a2);
    }
    __syncthreads();
    rb[tid] = (tid<DD) ? pa_w2[tid]*hpa[tid] : 0.f;
    __syncthreads();
    for (int off=128; off; off>>=1){ if(tid<off) rb[tid]+=rb[tid+off]; __syncthreads(); }
    if (tid==0){
      float ab=sgm(rb[0]+pa_b2[0]);
      out[bs]=sgm(5.f*(ab-diff));
    }
    __syncthreads();

    // ---- to_get + commit all_state ----
    if (tid<DD){
      float acc=no_b2[tid];
      for (int j=0;j<DD;j++) acc += t1[j]*no2T[j*DD+tid];
      tg[tid]=tanhf(acc);
      as_[tid]=asv;
    }
    __syncthreads();

    // ---- second attention + skill-state additive update + last_time ----
    { float s=0.f; const float* row=ss+tid*DD;
      for (int d=0;d<DD;d++) s += tg[d]*row[d];
      s *= (1.f/DD);
      if (!npk[tid]) s=-1e9f;
      sc[tid]=s; }
    __syncthreads();
    rb[tid]=sc[tid]; __syncthreads();
    for (int off=128; off; off>>=1){ if(tid<off) rb[tid]=fmaxf(rb[tid],rb[tid+off]); __syncthreads(); }
    { float mx2=rb[0]; __syncthreads();
      float e=__expf(sc[tid]-mx2); sc[tid]=e; rb[tid]=e; }
    __syncthreads();
    for (int off=128; off; off>>=1){ if(tid<off) rb[tid]+=rb[tid+off]; __syncthreads(); }
    { float inv2=1.f/rb[0]; __syncthreads();
      float a2v=sc[tid]*inv2;
      float* row=ss+tid*DD;
      for (int d=0;d<DD;d++) row[d] += a2v*tg[d];
      if (npk[tid]) ltm[tid]=t; }
    __syncthreads();
  }
}

// ---------------------------------------------------------------------------
extern "C" void kernel_launch(void* const* d_in, const int* in_sizes, int n_in,
                              void* d_out, int out_size, void* d_ws, size_t ws_size,
                              hipStream_t stream)
{
  (void)in_sizes; (void)n_in; (void)out_size; (void)ws_size;
  const float* pro_embed   =(const float*)d_in[0];
  const float* skill_embed =(const float*)d_in[1];
  const float* skill_state0=(const float*)d_in[2];
  const float* time_state  =(const float*)d_in[3];
  const float* all_state0  =(const float*)d_in[4];
  const float* ans_emb     =(const float*)d_in[5];
  const float* pro_diff_w  =(const float*)d_in[6];
  const float* pl_w=(const float*)d_in[7];  const float* pl_b=(const float*)d_in[8];
  const float* sl_w=(const float*)d_in[9];  const float* sl_b=(const float*)d_in[10];
  const float* pc_w=(const float*)d_in[11]; const float* pc_b=(const float*)d_in[12];
  const float* pr_w=(const float*)d_in[13]; const float* pr_b=(const float*)d_in[14];
  const float* af_w1=(const float*)d_in[15];const float* af_b1=(const float*)d_in[16];
  const float* af_w2=(const float*)d_in[17];const float* af_b2=(const float*)d_in[18];
  const float* sf_w1=(const float*)d_in[19];const float* sf_b1=(const float*)d_in[20];
  const float* sf_w2=(const float*)d_in[21];const float* sf_b2=(const float*)d_in[22];
  const float* no_w1=(const float*)d_in[23];const float* no_b1=(const float*)d_in[24];
  const float* no_w2=(const float*)d_in[25];const float* no_b2=(const float*)d_in[26];
  const float* pa_w1=(const float*)d_in[27];const float* pa_b1=(const float*)d_in[28];
  const float* pa_w2=(const float*)d_in[29];const float* pa_b2=(const float*)d_in[30];
  const float* ao_w=(const float*)d_in[31]; const float* ao_b=(const float*)d_in[32];
  const int* pro2skill   =(const int*)d_in[33];
  const int* next_problem=(const int*)d_in[36];
  const int* next_ans    =(const int*)d_in[37];
  float* out=(float*)d_out;

  char* wp=(char*)d_ws;
  auto alloc=[&](size_t bytes)->char*{ char* p=wp; wp += (bytes + 255) & ~(size_t)255; return p; };
  float* afw1T=(float*)alloc((size_t)DD*2*DD*4);
  float* afw2T=(float*)alloc((size_t)DD*DD*4);
  float* sfcT =(float*)alloc((size_t)DD*DD*4);
  float* prT  =(float*)alloc((size_t)DD*3*DD*4);
  float* paT  =(float*)alloc((size_t)DD*3*DD*4);
  float* no1T =(float*)alloc((size_t)DD*DD*4);
  float* no2T =(float*)alloc((size_t)DD*DD*4);
  float* aoT  =(float*)alloc((size_t)DD*DD*4);
  float* plT  =(float*)alloc((size_t)DD*DD*4);
  float* pcT  =(float*)alloc((size_t)DD*DD*4);
  bf16*  w1t  =(bf16*) alloc((size_t)DD*DD*2);
  bf16*  w2t  =(bf16*) alloc((size_t)DD*DD*2);
  bf16*  tp   =(bf16*) alloc((size_t)SL*DD*2);
  float* q_skill =(float*)alloc((size_t)KD*DD*4);
  float* pro_e   =(float*)alloc((size_t)PMAX*DD*4);
  float* diff_sig=(float*)alloc((size_t)PMAX*4);
  float* rasch   =(float*)alloc((size_t)BB*SL*DD*4);
  float* Xg      =(float*)alloc((size_t)BB*SL*DD*4);
  float* diff_bs =(float*)alloc((size_t)BB*SL*4);

  k_prep<<<256,256,0,stream>>>(af_w1,af_w2,sf_w1,sf_w2,pr_w,pa_w1,no_w1,no_w2,ao_w,pl_w,pc_w,
                               afw1T,afw2T,sfcT,prT,paT,no1T,no2T,aoT,plT,pcT,w1t,w2t);
  k_timeproj<<<SL,DD,0,stream>>>(time_state, sf_w1, tp);
  k_qskill<<<KD,DD,0,stream>>>(skill_embed, sl_w, sl_b, q_skill);
  k_pro<<<PMAX,DD,0,stream>>>(pro_embed, skill_embed, pl_b, pc_b, pro_diff_w, pro2skill,
                              plT, pcT, q_skill, pro_e, diff_sig);
  k_gather<<<BB*SL,DD,0,stream>>>(next_problem, next_ans, pro_e, ans_emb, diff_sig,
                                  rasch, Xg, diff_bs);

  size_t smem = 262144 + (size_t)4*3104;   // ss + m1 + w1 + w2 + vectors (~274.5KB)
  k_scan<<<BB,256,smem,stream>>>(skill_state0, all_state0, time_state,
                                 af_b1, af_b2, sf_b1, sf_b2, pr_b, pa_b1, pa_w2, pa_b2,
                                 no_b1, no_b2, ao_b,
                                 afw1T, afw2T, sfcT, prT, paT, no1T, no2T, aoT,
                                 w1t, w2t, tp,
                                 pro2skill, next_problem, rasch, Xg, diff_bs, out);
}
